// WindowAttention_34866544509740
// MI455X (gfx1250) — matrix operand order repaired and checked by reading, not compiled, not run on hardware
//
#include <hip/hip_runtime.h>

typedef unsigned short u16;
typedef unsigned int   u32;
typedef __bf16 bf16;
typedef bf16  v16bf __attribute__((ext_vector_type(16)));
typedef float v8f   __attribute__((ext_vector_type(8)));

#define DIM   256
#define HEADS 8
#define TOK   64
#define HD    32
#define BWIN  4096
#define NW    64

// ---------------- LDS layout (u16 units) ----------------
#define XS      272                       // xb / outb row stride (pad 16 -> 8-bank shift/row)
#define XB_OFF  0                         // 64*272 = 17408 shorts (reused as outb)
#define QK_STRD 40                        // q/k row stride (pad 8)
#define AT_STRD 80                        // attn row stride
#define VT_STRD 72                        // v^T row stride (pad 8)
#define QK_OFF  17408                     // per head: q 64*40 + k 64*40 = 5120 (overlaid by attnS 64*80)
#define VT_OFF  (17408 + 8*5120)          // per head: 32*72 = 2304
#define SMEM_U16 (17408 + 8*5120 + 8*2304)   // 76800 shorts = 153600 B -> 2 WG / WGP

// ---------------- bf16 helpers: native converts (v_cvt_*bf16* on gfx1250) ----------------
static __device__ inline u16 f2bf(float f) {
  bf16 h = (bf16)f;                       // RNE hardware convert
  return __builtin_bit_cast(u16, h);
}
static __device__ inline u32 pack2(float lo, float hi) {
  return (u32)f2bf(lo) | ((u32)f2bf(hi) << 16);
}

static __device__ inline v8f v8f_splat(float f) {
  v8f v;
#pragma unroll
  for (int i = 0; i < 8; ++i) v[i] = f;
  return v;
}

union FragU { v16bf v; uint4 q[2]; };

// A operand (16x32, M x K): source row-major [m][k].  Lane m = lane&15,
// k-chunks at {0..7,16..23} (lanes 0-15) / {8..15,24..31} (lanes 16-31).
static __device__ inline v16bf ld_frag_a(const u16* base, int stride, int lane) {
  const u16* p = base + (lane & 15) * stride + ((lane < 16) ? 0 : 8);
  FragU u;
  u.q[0] = *(const uint4*)(p);
  u.q[1] = *(const uint4*)(p + 16);
  return u.v;
}

// B operand (32x16, K x N): source stored transposed [n][k] (k contiguous).
// Lane n = lane&15, k-chunks at {0..7,16..23} (lanes 0-15) / {8..15,24..31} (lanes 16-31).
static __device__ inline v16bf ld_frag_b(const u16* baseT, int stride, int lane) {
  const u16* p = baseT + (lane & 15) * stride + ((lane < 16) ? 0 : 8);
  FragU u;
  u.q[0] = *(const uint4*)(p);
  u.q[1] = *(const uint4*)(p + 16);
  return u.v;
}

// store 8 f32 accumulator values as 8 contiguous bf16 (16B)
static __device__ inline void st_bf8(u16* p, v8f a) {
  uint4 o = { pack2(a[0], a[1]), pack2(a[2], a[3]), pack2(a[4], a[5]), pack2(a[6], a[7]) };
  *(uint4*)p = o;
}

#define WMMA_BF16(A, B, C) \
  __builtin_amdgcn_wmma_f32_16x16x32_bf16(false, (A), false, (B), (short)0, (C), false, false)

// ---------------- prep kernels (run once, results stay L2-resident) ----------------
__global__ void prep_weights(const float* __restrict__ qkvw, const float* __restrict__ projw,
                             u16* __restrict__ wqkv, u16* __restrict__ wproj) {
  int i = blockIdx.x * blockDim.x + threadIdx.x;
  int stride = gridDim.x * blockDim.x;
  for (int idx = i; idx < 768 * 256; idx += stride) wqkv[idx] = f2bf(qkvw[idx]);
  for (int idx = i; idx < 256 * 256; idx += stride) wproj[idx] = f2bf(projw[idx]);
}

__global__ void prep_bias(const float* __restrict__ bias_table, const int* __restrict__ rel_idx,
                          float* __restrict__ biasH) {
  int i = blockIdx.x * blockDim.x + threadIdx.x;  // [h][n][m]
  if (i < HEADS * TOK * TOK) {
    int h = i >> 12, nm = i & 4095;
    biasH[i] = bias_table[rel_idx[nm] * HEADS + h];
  }
}

// ---------------- main fused window-attention kernel ----------------
__global__ __launch_bounds__(256) void win_attn(
    const float* __restrict__ x, const float* __restrict__ mask,
    const float* __restrict__ qkv_b, const float* __restrict__ proj_b,
    const u16* __restrict__ wqkv, const u16* __restrict__ wproj,
    const float* __restrict__ biasH, float* __restrict__ out) {
  extern __shared__ u16 smem[];
  const int b    = blockIdx.x;
  const int tid  = threadIdx.x;
  const int lane = tid & 31;
  const int h    = tid >> 5;                 // wave = head
  const int col  = lane & 15;                // owned N-column in B/C/D fragments
  const int hi8  = (lane < 16) ? 0 : 8;      // M-row offset of this half-wave in C/D

  u16* xb  = smem + XB_OFF;                  // [64][XS] bf16 x, later reused as outb
  u16* qs  = smem + QK_OFF + h * 5120;       // [64][40] q (scaled, +bias)
  u16* ks  = qs + 2560;                      // [64][40] k (+bias)
  u16* ats = qs;                             // [64][80] softmaxed attn, overlays q+k
  u16* vts = smem + VT_OFF + h * 2304;       // [32][72] v^T (+bias)

  // ---- Phase 0: stage x[b] (64x256 f32) into LDS as bf16 ----
  const float* xg = x + (size_t)b * TOK * DIM;
  for (int i = tid; i < TOK * DIM / 4; i += 256) {
    int row = i >> 6, c4 = (i & 63) * 4;
    float4 f = *(const float4*)(xg + row * DIM + c4);
    *(uint2*)(xb + row * XS + c4) = make_uint2(pack2(f.x, f.y), pack2(f.z, f.w));
  }
  __syncthreads();

  // ---- Phase 1: qkv^T[j][n] = sum_c W[j][c] * x[n][c]  (per-head 96x64, K=256)
  //      2 groups of 3 feature-tiles so each x-fragment is loaded once per group ----
  const float scale = 0.1767766952966369f;   // 1/sqrt(32)
#pragma unroll
  for (int g = 0; g < 2; ++g) {
    v8f acc[3][4];
#pragma unroll
    for (int fi = 0; fi < 3; ++fi)
#pragma unroll
      for (int nt = 0; nt < 4; ++nt) acc[fi][nt] = v8f_splat(0.0f);

#pragma unroll
    for (int kk = 0; kk < 8; ++kk) {
      v16bf bf[4];
#pragma unroll
      for (int nt = 0; nt < 4; ++nt)
        bf[nt] = ld_frag_b(xb + nt * 16 * XS + kk * 32, XS, lane);
#pragma unroll
      for (int fi = 0; fi < 3; ++fi) {
        int f  = g * 3 + fi;                 // 0,1:q  2,3:k  4,5:v
        int j0 = (f < 2) ? (h * 32 + f * 16)
               : (f < 4) ? (256 + h * 32 + (f - 2) * 16)
                         : (512 + h * 32 + (f - 4) * 16);
        v16bf a = ld_frag_a(wqkv + j0 * DIM + kk * 32, DIM, lane);   // weights (L2-hot)
#pragma unroll
        for (int nt = 0; nt < 4; ++nt)
          acc[fi][nt] = WMMA_BF16(a, bf[nt], acc[fi][nt]);
      }
    }

#pragma unroll
    for (int fi = 0; fi < 3; ++fi) {
      int f  = g * 3 + fi;
      int j0 = (f < 2) ? (h * 32 + f * 16)
             : (f < 4) ? (256 + h * 32 + (f - 2) * 16)
                       : (512 + h * 32 + (f - 4) * 16);
      float bj[8];
#pragma unroll
      for (int r = 0; r < 8; ++r) bj[r] = qkv_b[j0 + hi8 + r];
#pragma unroll
      for (int nt = 0; nt < 4; ++nt) {
        int n = nt * 16 + col;               // token owned by this lane
        if (f < 2) {                         // q: (acc+bias)*scale -> qs[n][d]
          v8f t;
#pragma unroll
          for (int r = 0; r < 8; ++r) t[r] = (acc[fi][nt][r] + bj[r]) * scale;
          st_bf8(qs + n * QK_STRD + (f & 1) * 16 + hi8, t);
        } else if (f < 4) {                  // k -> ks[n][d]
          v8f t;
#pragma unroll
          for (int r = 0; r < 8; ++r) t[r] = acc[fi][nt][r] + bj[r];
          st_bf8(ks + n * QK_STRD + (f & 1) * 16 + hi8, t);
        } else {                             // v -> v^T[d][m] (scattered b16)
#pragma unroll
          for (int r = 0; r < 8; ++r)
            vts[((f & 1) * 16 + hi8 + r) * VT_STRD + n] = f2bf(acc[fi][nt][r] + bj[r]);
        }
      }
    }
  }
  __syncthreads();                           // xb free for reuse as outb

  // ---- Phase 2: attn^T[m][n] = k . q^T, C preloaded with bias+mask ----
  const int w = b & (NW - 1);
  v8f sacc[4][4];                            // [mt][nt]
  {
    const float* bh = biasH + h * 4096;
    const float* mk = mask + w * 4096;
#pragma unroll
    for (int nt = 0; nt < 4; ++nt) {
      int n = nt * 16 + col;
#pragma unroll
      for (int mt = 0; mt < 4; ++mt) {
        int mb = mt * 16 + hi8;
        const float4* bp = (const float4*)(bh + n * 64 + mb);
        const float4* mp = (const float4*)(mk + n * 64 + mb);
        float4 b0 = bp[0], b1 = bp[1], m0 = mp[0], m1 = mp[1];
        v8f c;
        c[0] = b0.x + m0.x; c[1] = b0.y + m0.y; c[2] = b0.z + m0.z; c[3] = b0.w + m0.w;
        c[4] = b1.x + m1.x; c[5] = b1.y + m1.y; c[6] = b1.z + m1.z; c[7] = b1.w + m1.w;
        sacc[mt][nt] = c;
      }
    }
    v16bf afr[4], bfr[4];
#pragma unroll
    for (int mt = 0; mt < 4; ++mt) afr[mt] = ld_frag_a(ks + mt * 16 * QK_STRD, QK_STRD, lane);
#pragma unroll
    for (int nt = 0; nt < 4; ++nt) bfr[nt] = ld_frag_b(qs + nt * 16 * QK_STRD, QK_STRD, lane);
#pragma unroll
    for (int mt = 0; mt < 4; ++mt)
#pragma unroll
      for (int nt = 0; nt < 4; ++nt)
        sacc[mt][nt] = WMMA_BF16(afr[mt], bfr[nt], sacc[mt][nt]);   // K = head_dim = 32
  }

  // ---- Phase 3: softmax over m (32 values in-lane + partner half-wave) ----
#pragma unroll
  for (int nt = 0; nt < 4; ++nt) {
    int n = nt * 16 + col;
    float mx = -3.0e38f;
#pragma unroll
    for (int mt = 0; mt < 4; ++mt)
#pragma unroll
      for (int r = 0; r < 8; ++r) mx = fmaxf(mx, sacc[mt][nt][r]);
    mx = fmaxf(mx, __shfl_xor(mx, 16, 32));
    float s = 0.0f;
#pragma unroll
    for (int mt = 0; mt < 4; ++mt)
#pragma unroll
      for (int r = 0; r < 8; ++r) {
        float e = __expf(sacc[mt][nt][r] - mx);
        sacc[mt][nt][r] = e;
        s += e;
      }
    s += __shfl_xor(s, 16, 32);
    float inv = 1.0f / s;
#pragma unroll
    for (int mt = 0; mt < 4; ++mt) {
      v8f t;
#pragma unroll
      for (int r = 0; r < 8; ++r) t[r] = sacc[mt][nt][r] * inv;
      st_bf8(ats + n * AT_STRD + mt * 16 + hi8, t);   // attn[n][m], m contiguous
    }
  }

  // ---- Phase 4: out[n][d] = attn . v  (A = attnS[n][m], B src = v^T[d][m]);
  //      all 12 fragments hoisted, 16 WMMAs back-to-back ----
  {
    v16bf aat[4][2], bvt[2][2];
#pragma unroll
    for (int nt = 0; nt < 4; ++nt) {
      aat[nt][0] = ld_frag_a(ats + nt * 16 * AT_STRD + 0,  AT_STRD, lane);
      aat[nt][1] = ld_frag_a(ats + nt * 16 * AT_STRD + 32, AT_STRD, lane);
    }
#pragma unroll
    for (int dt = 0; dt < 2; ++dt) {
      bvt[dt][0] = ld_frag_b(vts + dt * 16 * VT_STRD + 0,  VT_STRD, lane);
      bvt[dt][1] = ld_frag_b(vts + dt * 16 * VT_STRD + 32, VT_STRD, lane);
    }
#pragma unroll
    for (int nt = 0; nt < 4; ++nt)
#pragma unroll
      for (int dt = 0; dt < 2; ++dt) {
        v8f acc = v8f_splat(0.0f);
        acc = WMMA_BF16(aat[nt][0], bvt[dt][0], acc);
        acc = WMMA_BF16(aat[nt][1], bvt[dt][1], acc);
        int c = h * 32 + dt * 16 + col;
#pragma unroll
        for (int r = 0; r < 8; ++r) {
          int n = nt * 16 + hi8 + r;
          xb[n * XS + c] = f2bf(acc[r]);     // outb overlays xb
        }
      }
  }
  __syncthreads();                           // all heads' outb columns ready

  // ---- Phase 5: final[n][e] = out . proj_w^T + proj_b  (wave owns 32 e-cols) ----
  const int e0 = h * 32;
  float* og = out + (size_t)b * TOK * DIM;
#pragma unroll
  for (int et = 0; et < 2; ++et) {
    v16bf bfr[8];
#pragma unroll
    for (int kk = 0; kk < 8; ++kk)
      bfr[kk] = ld_frag_b(wproj + (e0 + et * 16) * DIM + kk * 32, DIM, lane);
    int e = e0 + et * 16 + col;
    float pb = proj_b[e];
#pragma unroll
    for (int nt = 0; nt < 4; ++nt) {
      v8f acc = v8f_splat(pb);
#pragma unroll
      for (int kk = 0; kk < 8; ++kk) {
        v16bf a = ld_frag_a(xb + nt * 16 * XS + kk * 32, XS, lane);
        acc = WMMA_BF16(a, bfr[kk], acc);
      }
#pragma unroll
      for (int r = 0; r < 8; ++r)
        og[(nt * 16 + hi8 + r) * DIM + e] = acc[r];
    }
  }
}

// ---------------- host launcher ----------------
extern "C" void kernel_launch(void* const* d_in, const int* in_sizes, int n_in,
                              void* d_out, int out_size, void* d_ws, size_t ws_size,
                              hipStream_t stream) {
  const float* x    = (const float*)d_in[0];
  const float* mask = (const float*)d_in[1];
  const float* qkvw = (const float*)d_in[2];
  const float* qkvb = (const float*)d_in[3];
  const float* pw   = (const float*)d_in[4];
  const float* pb   = (const float*)d_in[5];
  const float* bt   = (const float*)d_in[6];
  const int*   ridx = (const int*)d_in[7];

  u16*   wqkv  = (u16*)d_ws;                                       // 768*256*2  = 384 KB
  u16*   wproj = (u16*)((char*)d_ws + 768 * 256 * 2);              // 256*256*2  = 128 KB
  float* biasH = (float*)((char*)d_ws + 768 * 256 * 2 + 256 * 256 * 2); // 8*64*64*4 = 128 KB

  prep_weights<<<768, 256, 0, stream>>>(qkvw, pw, wqkv, wproj);
  prep_bias<<<128, 256, 0, stream>>>(bt, ridx, biasH);
  win_attn<<<BWIN, 256, SMEM_U16 * 2, stream>>>(x, mask, qkvb, pb, wqkv, wproj, biasH,
                                                (float*)d_out);
}
